// FCN_81913616269757
// MI455X (gfx1250) — matrix operand order repaired and checked by reading, not compiled, not run on hardware
//
#include <hip/hip_runtime.h>

typedef __attribute__((ext_vector_type(16))) __bf16 v16bf;
typedef __attribute__((ext_vector_type(8)))  __bf16 v8bf;
typedef __attribute__((ext_vector_type(8)))  float  v8f;
typedef __attribute__((ext_vector_type(4)))  float  v4f;

__device__ __forceinline__ float fast_tanh(float v) {
#if __has_builtin(__builtin_amdgcn_tanhf)
    return __builtin_amdgcn_tanhf(v);   // v_tanh_f32 on gfx1250
#else
    return tanhf(v);
#endif
}

// Pin a v8f in VGPRs and make its contents opaque to the optimizer so the
// loop-invariant WMMA C operands are never re-materialized per iteration.
__device__ __forceinline__ void pin_v8f(v8f& v) {
    asm volatile("" : "+v"(v));
}

// After a WMMA layer, each lane holds ha = h[lane%16] and hb = h[16 + lane%16]
// (D VGPR0 broadcasts y[0..15] / y[16..31] across both lane halves).
// Write h (bf16) to LDS, reload as the 16x32 A operand in this file's contiguous k order (same as load_Bblk):
//   lanes 0-15 : elems 0-15 -> K=0..15
//   lanes 16-31: elems 0-15 -> K=16..31
// Every lane gets the same chunk -> every A row equals h.
__device__ __forceinline__ v16bf marshal_h(float ha, float hb,
                                           __bf16* buf, int l16, int g) {
    buf[l16]      = (__bf16)ha;   // lanes L and L+16 write identical data
    buf[16 + l16] = (__bf16)hb;
    asm volatile("s_wait_dscnt 0" ::: "memory");
    const v8bf* p = (const v8bf*)buf;
    v8bf lo = p[g ? 2 : 0];   // ds_load_b128
    v8bf hi = p[g ? 3 : 1];
    v16bf a;
#pragma unroll
    for (int i = 0; i < 8; ++i) { a[i] = lo[i]; a[8 + i] = hi[i]; }
    return a;
}

// B-matrix block (32x16 bf16) holding W^T: B[k][n] = W[n][k].
// Lane layout: N = lane%16; lanes 0-15 cover K=0..15, lanes 16-31 cover K=16..31.
__device__ __forceinline__ v16bf load_Bblk(const float* __restrict__ W,
                                           int ncol, int g) {
    v16bf b;
    const int kb = g ? 16 : 0;
#pragma unroll
    for (int i = 0; i < 16; ++i) b[i] = (__bf16)W[ncol * 32 + kb + i];
    return b;
}

__global__ __launch_bounds__(32) void rnn3_scan_kernel(
    const float* __restrict__ x,
    const float* __restrict__ x_lb, const float* __restrict__ x_ub,
    const float* __restrict__ W_ih0, const float* __restrict__ W_hh0,
    const float* __restrict__ b_ih0, const float* __restrict__ b_hh0,
    const float* __restrict__ W_ih1, const float* __restrict__ W_hh1,
    const float* __restrict__ b_ih1, const float* __restrict__ b_hh1,
    const float* __restrict__ Wp_ih, const float* __restrict__ Wp_hh,
    const float* __restrict__ bp_ih, const float* __restrict__ bp_hh,
    const float* __restrict__ prev_h0, const float* __restrict__ post_h0,
    float* __restrict__ out, int T)
{
    __shared__ __bf16 h0buf[32];
    __shared__ __bf16 h1buf[32];

    const int lane = threadIdx.x;
    const int l16  = lane & 15;
    const int g    = lane >> 4;
    const int kb   = g ? 16 : 0;

    // Weight blocks W^T as WMMA B operands (VGPR-resident for the whole scan).
    v16bf Bh0_0 = load_Bblk(W_hh0, l16,      g);
    v16bf Bh0_1 = load_Bblk(W_hh0, 16 + l16, g);
    v16bf Bi1_0 = load_Bblk(W_ih1, l16,      g);
    v16bf Bi1_1 = load_Bblk(W_ih1, 16 + l16, g);
    v16bf Bh1_0 = load_Bblk(W_hh1, l16,      g);
    v16bf Bh1_1 = load_Bblk(W_hh1, 16 + l16, g);

    // Head weights as a B block: column 0 = Wp_ih, other columns zero.
    v16bf Bwp;
#pragma unroll
    for (int i = 0; i < 16; ++i)
        Bwp[i] = (l16 == 0) ? (__bf16)Wp_ih[kb + i] : (__bf16)0.0f;

    // Per-lane input-weight constants (applied AFTER the WMMA, before tanh).
    const float wx0a = W_ih0[l16];
    const float wx0b = W_ih0[16 + l16];

    const float xlb  = x_lb[0];
    const float xinv = 1.0f / (x_ub[0] - xlb);
    const float wph  = Wp_hh[0];
    float hp         = post_h0[0];

    // Initial hidden states into A-layout.
    v16bf A0 = marshal_h(prev_h0[l16],      prev_h0[16 + l16], h0buf, l16, g);
    v16bf A1 = marshal_h(prev_h0[32 + l16], prev_h0[48 + l16], h1buf, l16, g);

    // ALL C operands are loop-invariant (bias-only); pin them in VGPRs.
    v8f C0a = {}; C0a[0] = b_ih0[l16]      + b_hh0[l16];
    v8f C0b = {}; C0b[0] = b_ih0[16 + l16] + b_hh0[16 + l16];
    v8f C1a = {}; C1a[0] = b_ih1[l16]      + b_hh1[l16];
    v8f C1b = {}; C1b[0] = b_ih1[16 + l16] + b_hh1[16 + l16];
    v8f Cp  = {}; Cp[0]  = bp_ih[0] + bp_hh[0];
    pin_v8f(C0a); pin_v8f(C0b); pin_v8f(C1a); pin_v8f(C1b); pin_v8f(Cp);

    // Software-pipelined uniform x loads: fetch chunk tb+8 while processing tb.
    v4f xA = *(const v4f*)(x);
    v4f xB = *(const v4f*)(x + 4);

    for (int tb = 0; tb < T; tb += 8) {
        const int tn = (tb + 8 < T) ? (tb + 8) : tb;   // clamped prefetch addr
        v4f nA = *(const v4f*)(x + tn);                // issued now, waited on
        v4f nB = *(const v4f*)(x + tn + 4);            // ~8 steps later

        float xv[8] = { xA[0], xA[1], xA[2], xA[3], xB[0], xB[1], xB[2], xB[3] };
        float hv[8];

#pragma unroll
        for (int i = 0; i < 8; ++i) {
            const float xn = (xv[i] - xlb) * xinv;      // off the critical path

            // ---- layer 0: h0 = tanh(W_hh0 h0 + b + W_ih0 xn) ----
            v8f d0 = __builtin_amdgcn_wmma_f32_16x16x32_bf16(false, A0, false, Bh0_0, (short)0, C0a, false, false);
            v8f d1 = __builtin_amdgcn_wmma_f32_16x16x32_bf16(false, A0, false, Bh0_1, (short)0, C0b, false, false);
            const float h0a = fast_tanh(fmaf(wx0a, xn, d0[0]));
            const float h0b = fast_tanh(fmaf(wx0b, xn, d1[0]));
            A0 = marshal_h(h0a, h0b, h0buf, l16, g);

            // ---- layer 1: h1 = tanh(W_ih1 h0 + W_hh1 h1 + b) (accumulating WMMAs) ----
            v8f e0 = __builtin_amdgcn_wmma_f32_16x16x32_bf16(false, A1, false, Bh1_0, (short)0, C1a, false, false);
            v8f e1 = __builtin_amdgcn_wmma_f32_16x16x32_bf16(false, A1, false, Bh1_1, (short)0, C1b, false, false);
            e0 = __builtin_amdgcn_wmma_f32_16x16x32_bf16(false, A0, false, Bi1_0, (short)0, e0, false, false);
            e1 = __builtin_amdgcn_wmma_f32_16x16x32_bf16(false, A0, false, Bi1_1, (short)0, e1, false, false);
            const float h1a = fast_tanh(e0[0]);
            const float h1b = fast_tanh(e1[0]);
            A1 = marshal_h(h1a, h1b, h1buf, l16, g);

            // ---- head: D[0][0] = Wp_ih . h1 + pb lands in lane 0, VGPR0 ----
            v8f f0 = __builtin_amdgcn_wmma_f32_16x16x32_bf16(false, A1, false, Bwp, (short)0, Cp, false, false);
            hp = fast_tanh(fmaf(wph, hp, f0[0]));   // only lane 0's value is real
            hv[i] = hp;
        }

        // Single predicated region per 8 steps; two b128 stores from lane 0.
        if (lane == 0) {
            v4f o0 = { hv[0], hv[1], hv[2], hv[3] };
            v4f o1 = { hv[4], hv[5], hv[6], hv[7] };
            *(v4f*)(out + tb)     = o0;
            *(v4f*)(out + tb + 4) = o1;
        }

        xA = nA; xB = nB;
    }
}

extern "C" void kernel_launch(void* const* d_in, const int* in_sizes, int n_in,
                              void* d_out, int out_size, void* d_ws, size_t ws_size,
                              hipStream_t stream) {
    (void)n_in; (void)d_ws; (void)ws_size; (void)out_size;
    const float* x       = (const float*)d_in[0];
    const float* x_lb    = (const float*)d_in[1];
    const float* x_ub    = (const float*)d_in[2];
    const float* W_ih0   = (const float*)d_in[3];
    const float* W_hh0   = (const float*)d_in[4];
    const float* b_ih0   = (const float*)d_in[5];
    const float* b_hh0   = (const float*)d_in[6];
    const float* W_ih1   = (const float*)d_in[7];
    const float* W_hh1   = (const float*)d_in[8];
    const float* b_ih1   = (const float*)d_in[9];
    const float* b_hh1   = (const float*)d_in[10];
    const float* Wp_ih   = (const float*)d_in[11];
    const float* Wp_hh   = (const float*)d_in[12];
    const float* bp_ih   = (const float*)d_in[13];
    const float* bp_hh   = (const float*)d_in[14];
    const float* prev_h0 = (const float*)d_in[15];
    const float* post_h0 = (const float*)d_in[16];
    float* out = (float*)d_out;
    const int T = in_sizes[0];

    rnn3_scan_kernel<<<1, 32, 0, stream>>>(
        x, x_lb, x_ub, W_ih0, W_hh0, b_ih0, b_hh0,
        W_ih1, W_hh1, b_ih1, b_hh1,
        Wp_ih, Wp_hh, bp_ih, bp_hh, prev_h0, post_h0, out, T);
}
